// GCL_2774548873594
// MI455X (gfx1250) — matrix operand order repaired and checked by reading, not compiled, not run on hardware
//
#include <hip/hip_runtime.h>

typedef __attribute__((ext_vector_type(16))) __bf16 v16bf;
typedef __attribute__((ext_vector_type(8)))  float  v8f;

__device__ __forceinline__ __bf16 f2bf(float f) {
    // round-to-nearest-even f32 -> bf16
    unsigned u = __builtin_bit_cast(unsigned, f);
    unsigned r = u + 0x7FFFu + ((u >> 16) & 1u);
    unsigned short h = (unsigned short)(r >> 16);
    return __builtin_bit_cast(__bf16, h);
}

// ---------------------------------------------------------------------------
// Generic [nrows,K] x [K,128] + bias GEMM, bf16 WMMA with f32 accumulation.
// Block = 256 threads = 8 waves. Each wave computes one 16-row x 128-col tile.
// W is staged in LDS pre-swizzled into the WMMA B-operand lane layout:
//   B tile (32x16 bf16): lane l holds column n = l&15, K-half = l>>4,
//   element j (0..15) = W[kt*32 + 16*(l>>4) + j][nt*16 + (l&15)]
// A operand lane layout (16x32 bf16): lane l = row (l&15), K-half kh = l>>4,
//   elements 0..7  -> K = kt*32 + 16*kh + j
//   elements 8..15 -> K = kt*32 + 16*kh + 8 + j
// C/D (16x16 f32, 8 VGPRs): lane l -> col n = l&15, row M = r + 8*(l>>4)
// ---------------------------------------------------------------------------
__global__ __launch_bounds__(256) void gemm_wmma_bf16(
    const float* __restrict__ A, const float* __restrict__ W,
    const float* __restrict__ bias, float* __restrict__ out,
    int nrows, int K, int relu_in, int relu_out)
{
    extern __shared__ __bf16 sW[];           // [NKT][8 nt][32 lanes][16 elems]
    const int NKT  = K >> 5;                 // K-tiles of 32
    const int tid  = threadIdx.x;
    const int lane = tid & 31;
    const int wave = tid >> 5;

    // Cooperative fill of swizzled bf16 weights into LDS.
    const int total = NKT << 12;             // NKT * 8 * 32 * 16
    for (int idx = tid; idx < total; idx += 256) {
        int j  = idx & 15;
        int l  = (idx >> 4) & 31;
        int nt = (idx >> 9) & 7;
        int kt = idx >> 12;
        int k  = kt * 32 + 16 * (l >> 4) + j;
        int n  = nt * 16 + (l & 15);
        sW[idx] = f2bf(W[k * 128 + n]);
    }
    __syncthreads();

    const int tile = blockIdx.x * 8 + wave;
    if (tile * 16 >= nrows) return;          // wave-uniform: EXEC stays all-ones

    const int m  = lane & 15;
    const int kh = lane >> 4;
    const long long row = (long long)tile * 16 + m;

    v8f c[8];
#pragma unroll
    for (int i = 0; i < 8; ++i)
#pragma unroll
        for (int e = 0; e < 8; ++e) c[i][e] = 0.0f;

    for (int kt = 0; kt < NKT; ++kt) {
        const float* ap = A + row * K + kt * 32 + kh * 16;
        v16bf a;
#pragma unroll
        for (int j = 0; j < 8; ++j) {
            float v0 = ap[j];
            float v1 = ap[j + 8];
            if (relu_in) { v0 = fmaxf(v0, 0.0f); v1 = fmaxf(v1, 0.0f); }
            a[j]     = f2bf(v0);
            a[j + 8] = f2bf(v1);
        }
        const __bf16* bp = sW + (((kt * 8) * 32) + lane) * 16;
#pragma unroll
        for (int nt = 0; nt < 8; ++nt) {
            v16bf b = *(const v16bf*)(bp + nt * 512);   // 32 lanes * 16 elems
            c[nt] = __builtin_amdgcn_wmma_f32_16x16x32_bf16(
                        false, a, false, b, (short)0, c[nt], false, false);
        }
    }

    const int col = lane & 15;
    const long long obase = (long long)tile * 16 * 128;
#pragma unroll
    for (int nt = 0; nt < 8; ++nt) {
        float bv = bias[nt * 16 + col];
#pragma unroll
        for (int r = 0; r < 8; ++r) {
            int mrow = r + 8 * kh;
            float v = c[nt][r] + bv;
            if (relu_out) v = fmaxf(v, 0.0f);
            out[obase + (long long)mrow * 128 + nt * 16 + col] = v;
        }
    }
}

// agg[dst[e], :] += feat[src[e], :] * w[e]; 32 lanes/edge, float4 per lane.
__global__ __launch_bounds__(256) void scatter_edges(
    const float* __restrict__ feat, const int* __restrict__ src,
    const int* __restrict__ dst, const float* __restrict__ w,
    float* __restrict__ agg, int E)
{
    long long t = (long long)blockIdx.x * blockDim.x + threadIdx.x;
    int e = (int)(t >> 5);
    if (e >= E) return;
    int c = (int)(t & 31) * 4;
    float wt = w[e];
    const float4 v = *(const float4*)(feat + (size_t)src[e] * 128 + c);
    float* p = agg + (size_t)dst[e] * 128 + c;
    unsafeAtomicAdd(p + 0, v.x * wt);   // global_atomic_add_f32
    unsafeAtomicAdd(p + 1, v.y * wt);
    unsafeAtomicAdd(p + 2, v.z * wt);
    unsafeAtomicAdd(p + 3, v.w * wt);
}

__global__ __launch_bounds__(256) void zero_f32(float* __restrict__ p, int n) {
    int i = blockIdx.x * blockDim.x + threadIdx.x;
    if (i < n) p[i] = 0.0f;
}

extern "C" void kernel_launch(void* const* d_in, const int* in_sizes, int n_in,
                              void* d_out, int out_size, void* d_ws, size_t ws_size,
                              hipStream_t stream) {
    const float* x      = (const float*)d_in[0];
    const float* edge_w = (const float*)d_in[1];
    const float* W1  = (const float*)d_in[2];
    const float* b1  = (const float*)d_in[3];
    const float* W2  = (const float*)d_in[4];
    const float* b2  = (const float*)d_in[5];
    const float* Wp1 = (const float*)d_in[6];
    const float* bp1 = (const float*)d_in[7];
    const float* Wp2 = (const float*)d_in[8];
    const float* bp2 = (const float*)d_in[9];
    const int* esrc  = (const int*)d_in[10];
    const int* edst  = (const int*)d_in[11];

    const int N = in_sizes[0] / 256;   // 100000
    const int E = in_sizes[1];         // 1600000

    float* z    = (float*)d_out;                 // output 0: [N,128]
    float* emb  = z + (size_t)N * 128;           // output 1: [N,128]
    float* bufA = (float*)d_ws;                  // [N,128]
    float* bufB = bufA + (size_t)N * 128;        // [N,128]

    const int tiles   = (N + 15) / 16;
    const int gblocks = (tiles + 7) / 8;
    const int nElem   = N * 128;
    const int zblocks = (nElem + 255) / 256;
    const long long sthreads = (long long)E * 32;
    const int sblocks = (int)((sthreads + 255) / 256);
    dim3 blk(256);

    // h_lin = x @ W1 + b1                       (K=256 -> 64KB LDS of bf16 W)
    gemm_wmma_bf16<<<gblocks, blk, 8 * 8192, stream>>>(x, W1, b1, bufA, N, 256, 0, 0);
    // agg1 = scatter_sum(h_lin[src] * w -> dst)
    zero_f32<<<zblocks, blk, 0, stream>>>(bufB, nElem);
    scatter_edges<<<sblocks, blk, 0, stream>>>(bufA, esrc, edst, edge_w, bufB, E);
    // t2 = relu(agg1) @ W2 + b2                 (relu fused into A conversion)
    gemm_wmma_bf16<<<gblocks, blk, 4 * 8192, stream>>>(bufB, W2, b2, bufA, N, 128, 1, 0);
    // emb = scatter_sum(t2[src] * w -> dst)     (directly into d_out half 2)
    zero_f32<<<zblocks, blk, 0, stream>>>(emb, nElem);
    scatter_edges<<<sblocks, blk, 0, stream>>>(bufA, esrc, edst, edge_w, emb, E);
    // p = relu(emb @ Wp1 + bp1)
    gemm_wmma_bf16<<<gblocks, blk, 4 * 8192, stream>>>(emb, Wp1, bp1, bufB, N, 128, 0, 1);
    // z = p @ Wp2 + bp2
    gemm_wmma_bf16<<<gblocks, blk, 4 * 8192, stream>>>(bufB, Wp2, bp2, z, N, 128, 0, 0);
}
